// MambaLayer_87368224735808
// MI455X (gfx1250) — compile-verified, matrix operand order checked by reading, not run on hardware
//
#include <hip/hip_runtime.h>
#include <hip/hip_bf16.h>

typedef __attribute__((ext_vector_type(16))) _Float16 v16h;
typedef __attribute__((ext_vector_type(8)))  _Float16 v8h;
typedef __attribute__((ext_vector_type(8)))  float    v8f;

#define DIM     256
#define DINNER  512
#define DSTATE  16
#define DTRANK  16
#define BATCH   8
#define LSEQ    1024
#define MROWS   (BATCH * LSEQ)   // 8192

// ---------------- WMMA helpers ----------------

__device__ __forceinline__ v8f wmma32(v16h a, v16h b, v8f c) {
    // D = A(16x32 f16) x B(32x16 f16) + C(16x16 f32)
    return __builtin_amdgcn_wmma_f32_16x16x32_f16(
        false, a, false, b, (short)0, c, false, false);
}

// Load one 16x32 f16 fragment. `base` points at the tile's row 0 / k0 column.
// Lane layout (ISA 7.12.2, 16-bit A 16x32): lane&15 = row (or col for B),
// lane>>4 = g; halves [0..7] hold K=8g..8g+7, halves [8..15] hold K=16+8g..23+8g.
// Both runs are contiguous 16-byte chunks -> two b128 loads.
__device__ __forceinline__ v16h load_frag16(const _Float16* __restrict__ base,
                                            int ld, int k0, int lane) {
    int m = lane & 15;
    int g = lane >> 4;
    const _Float16* p = base + (size_t)m * ld + (k0 + 8 * g);
    v8h lo = *(const v8h*)(p);
    v8h hi = *(const v8h*)(p + 16);
    v16h f;
#pragma unroll
    for (int i = 0; i < 8; ++i) { f[i] = lo[i]; f[i + 8] = hi[i]; }
    return f;
}

__device__ __forceinline__ float silu_f(float v) {
    return v / (1.f + __expf(-v));
}

// ---------------- prep kernels ----------------

// W_fused[n][c] = sum_o in_proj_w[n][o] * proj_w[o][c]  (f16)
// b_fused[n]   = sum_o in_proj_w[n][o] * proj_b[o]      (f32)
__global__ void fuse_w_kernel(const float* __restrict__ in_proj_w,
                              const float* __restrict__ proj_w,
                              const float* __restrict__ proj_b,
                              _Float16* __restrict__ Wf,
                              float* __restrict__ bf) {
    int idx = blockIdx.x * blockDim.x + threadIdx.x;   // 1024*256
    int n = idx >> 8, c = idx & 255;
    float s = 0.f;
    for (int o = 0; o < DIM; ++o)
        s += in_proj_w[n * DIM + o] * proj_w[o * DIM + c];
    Wf[n * DIM + c] = (_Float16)s;
    if (c == 0) {
        float t = 0.f;
        for (int o = 0; o < DIM; ++o)
            t += in_proj_w[n * DIM + o] * proj_b[o];
        bf[n] = t;
    }
}

// convert small weights to f16 (pad dt_proj K 16->32), A = -exp(A_log)
#define CSEG0 (48 * DINNER)        // x_proj_w
#define CSEG1 (DINNER * 32)        // dt_proj_w padded
#define CSEG2 (DIM * DINNER)       // out_proj_w
#define CSEG3 (DINNER * DSTATE)    // A
#define CTOT  (CSEG0 + CSEG1 + CSEG2 + CSEG3)
__global__ void convert_kernel(const float* __restrict__ x_proj_w,
                               const float* __restrict__ dt_proj_w,
                               const float* __restrict__ out_proj_w,
                               const float* __restrict__ A_log,
                               _Float16* __restrict__ xproj16,
                               _Float16* __restrict__ dtproj16,
                               _Float16* __restrict__ outproj16,
                               float* __restrict__ Afl) {
    int i = blockIdx.x * blockDim.x + threadIdx.x;
    if (i < CSEG0) {
        xproj16[i] = (_Float16)x_proj_w[i];
    } else if (i < CSEG0 + CSEG1) {
        int j = i - CSEG0, n = j >> 5, k = j & 31;
        dtproj16[j] = (k < DTRANK) ? (_Float16)dt_proj_w[n * DTRANK + k]
                                   : (_Float16)0.f;
    } else if (i < CSEG0 + CSEG1 + CSEG2) {
        int j = i - CSEG0 - CSEG1;
        outproj16[j] = (_Float16)out_proj_w[j];
    } else if (i < CTOT) {
        int j = i - CSEG0 - CSEG1 - CSEG2;
        Afl[j] = -__expf(A_log[j]);
    }
}

// xseq[b][l][c] = (f16) x[b][c][l]
__global__ void transpose_x_kernel(const float* __restrict__ x,
                                   _Float16* __restrict__ xseq) {
    int i = blockIdx.x * blockDim.x + threadIdx.x;   // B*DIM*LSEQ
    int b = i / (DIM * LSEQ);
    int r = i - b * DIM * LSEQ;
    int c = r / LSEQ, l = r - c * LSEQ;
    xseq[((size_t)(b * LSEQ + l)) * DIM + c] = (_Float16)x[i];
}

// ---------------- GEMM 1: xz = xseq @ Wf^T + bf  (M=8192,N=1024,K=256) ----------------

__global__ void gemm_xz_kernel(const _Float16* __restrict__ A,
                               const _Float16* __restrict__ W,
                               const float* __restrict__ bias,
                               _Float16* __restrict__ C) {
    const int N = 2 * DINNER, K = DIM;
    int wave = threadIdx.x >> 5, lane = threadIdx.x & 31;
    int macro = blockIdx.x * 8 + wave;
    const int mpr = N / 32;                 // macros per row = 32
    int m0 = (macro / mpr) * 32, n0 = (macro % mpr) * 32;
    const _Float16* A0 = A + (size_t)m0 * K;
    const _Float16* A1 = A + (size_t)(m0 + 16) * K;
    const _Float16* W0 = W + (size_t)n0 * K;
    const _Float16* W1 = W + (size_t)(n0 + 16) * K;
    v8f acc00 = {}, acc01 = {}, acc10 = {}, acc11 = {};
    for (int k0 = 0; k0 < K; k0 += 32) {
        if (k0 + 32 < K) {
            __builtin_prefetch(A0 + (size_t)(lane & 15) * K + k0 + 32, 0, 1);
            __builtin_prefetch(A1 + (size_t)(lane & 15) * K + k0 + 32, 0, 1);
        }
        v16h a0 = load_frag16(A0, K, k0, lane);
        v16h a1 = load_frag16(A1, K, k0, lane);
        v16h b0 = load_frag16(W0, K, k0, lane);
        v16h b1 = load_frag16(W1, K, k0, lane);
        acc00 = wmma32(a0, b0, acc00);
        acc01 = wmma32(a0, b1, acc01);
        acc10 = wmma32(a1, b0, acc10);
        acc11 = wmma32(a1, b1, acc11);
    }
    int g = lane >> 4, n = lane & 15;
#pragma unroll
    for (int r = 0; r < 8; ++r) {
        int mA = m0 + r + 8 * g, mB = m0 + 16 + r + 8 * g;
        int nA = n0 + n, nB = n0 + 16 + n;
        C[(size_t)mA * N + nA] = (_Float16)(acc00[r] + bias[nA]);
        C[(size_t)mA * N + nB] = (_Float16)(acc01[r] + bias[nB]);
        C[(size_t)mB * N + nA] = (_Float16)(acc10[r] + bias[nA]);
        C[(size_t)mB * N + nB] = (_Float16)(acc11[r] + bias[nB]);
    }
}

// ---------------- depthwise causal conv1d + silu ----------------

__global__ void conv_kernel(const _Float16* __restrict__ xz,
                            const float* __restrict__ cw,
                            const float* __restrict__ cb,
                            _Float16* __restrict__ xs) {
    int i = blockIdx.x * blockDim.x + threadIdx.x;   // B*L*DINNER
    int d = i & (DINNER - 1);
    int bl = i >> 9;                 // b*L + l
    int l = bl & (LSEQ - 1);
    int b = bl >> 10;
    float acc = cb[d];
#pragma unroll
    for (int k = 0; k < 4; ++k) {
        int ls = l - 3 + k;
        if (ls >= 0)
            acc += cw[d * 4 + k] *
                   (float)xz[((size_t)(b * LSEQ + ls)) * (2 * DINNER) + d];
    }
    xs[i] = (_Float16)silu_f(acc);
}

// ---------------- GEMM 2: x_dbl = xs @ x_proj_w^T  (M=8192,N=48,K=512) ----------------

__global__ void gemm_xdbl_kernel(const _Float16* __restrict__ A,
                                 const _Float16* __restrict__ W,
                                 _Float16* __restrict__ dtpad,   // 8192x32, hi zero
                                 float* __restrict__ Bm,         // 8192x16
                                 float* __restrict__ Cm) {       // 8192x16
    const int K = DINNER;
    int wave = threadIdx.x >> 5, lane = threadIdx.x & 31;
    int tile = blockIdx.x * 8 + wave;         // 1536 tiles
    int ntile = tile % 3, mtile = tile / 3;
    const _Float16* A0 = A + (size_t)(mtile * 16) * K;
    const _Float16* W0 = W + (size_t)(ntile * 16) * K;
    v8f acc = {};
    for (int k0 = 0; k0 < K; k0 += 32) {
        v16h a = load_frag16(A0, K, k0, lane);
        v16h b = load_frag16(W0, K, k0, lane);
        acc = wmma32(a, b, acc);
    }
    int g = lane >> 4, n = lane & 15;
#pragma unroll
    for (int r = 0; r < 8; ++r) {
        int m = mtile * 16 + r + 8 * g;
        if (ntile == 0) {
            dtpad[(size_t)m * 32 + n]      = (_Float16)acc[r];
            dtpad[(size_t)m * 32 + 16 + n] = (_Float16)0.f;
        } else if (ntile == 1) {
            Bm[(size_t)m * DSTATE + n] = acc[r];
        } else {
            Cm[(size_t)m * DSTATE + n] = acc[r];
        }
    }
}

// ---------------- GEMM 3: dt = softplus(dtpad @ dtproj^T + b)  (M=8192,N=512,K=32) ----------------

__global__ void gemm_dt_kernel(const _Float16* __restrict__ A,
                               const _Float16* __restrict__ W,
                               const float* __restrict__ bias,
                               float* __restrict__ dtv) {
    const int N = DINNER, K = 32;
    int wave = threadIdx.x >> 5, lane = threadIdx.x & 31;
    int macro = blockIdx.x * 8 + wave;
    const int mpr = N / 32;                 // 16
    int m0 = (macro / mpr) * 32, n0 = (macro % mpr) * 32;
    v16h a0 = load_frag16(A + (size_t)m0 * K, K, 0, lane);
    v16h a1 = load_frag16(A + (size_t)(m0 + 16) * K, K, 0, lane);
    v16h b0 = load_frag16(W + (size_t)n0 * K, K, 0, lane);
    v16h b1 = load_frag16(W + (size_t)(n0 + 16) * K, K, 0, lane);
    v8f acc00 = {}, acc01 = {}, acc10 = {}, acc11 = {};
    acc00 = wmma32(a0, b0, acc00);
    acc01 = wmma32(a0, b1, acc01);
    acc10 = wmma32(a1, b0, acc10);
    acc11 = wmma32(a1, b1, acc11);
    int g = lane >> 4, n = lane & 15;
#pragma unroll
    for (int r = 0; r < 8; ++r) {
        int mA = m0 + r + 8 * g, mB = m0 + 16 + r + 8 * g;
        int nA = n0 + n, nB = n0 + 16 + n;
        float v;
        v = acc00[r] + bias[nA]; dtv[(size_t)mA * N + nA] = (v > 20.f) ? v : log1pf(__expf(v));
        v = acc01[r] + bias[nB]; dtv[(size_t)mA * N + nB] = (v > 20.f) ? v : log1pf(__expf(v));
        v = acc10[r] + bias[nA]; dtv[(size_t)mB * N + nA] = (v > 20.f) ? v : log1pf(__expf(v));
        v = acc11[r] + bias[nB]; dtv[(size_t)mB * N + nB] = (v > 20.f) ? v : log1pf(__expf(v));
    }
}

// ---------------- selective scan (serial over L, parallel over B x d_inner) ----------------
// Fuses: h recurrence, y = h.C, + xs*D residual, * silu(z) gate -> yg (f16)

__global__ void scan_kernel(const float* __restrict__ dtv,
                            const float* __restrict__ Bm,
                            const float* __restrict__ Cm,
                            const _Float16* __restrict__ xs,
                            const _Float16* __restrict__ xz,
                            const float* __restrict__ Afl,
                            const float* __restrict__ Dv,
                            _Float16* __restrict__ yg) {
    int b = blockIdx.x >> 2;
    int d = ((blockIdx.x & 3) << 7) + threadIdx.x;   // 128 threads/block
    __shared__ float sB[DSTATE], sC[DSTATE];
    float h[DSTATE], Ad[DSTATE];
#pragma unroll
    for (int s = 0; s < DSTATE; ++s) { h[s] = 0.f; Ad[s] = Afl[d * DSTATE + s]; }
    float Dd = Dv[d];
    for (int t = 0; t < LSEQ; ++t) {
        size_t row = (size_t)(b * LSEQ + t);
        if (threadIdx.x < DSTATE)
            sB[threadIdx.x] = Bm[row * DSTATE + threadIdx.x];
        else if (threadIdx.x < 2 * DSTATE)
            sC[threadIdx.x - DSTATE] = Cm[row * DSTATE + threadIdx.x - DSTATE];
        __syncthreads();
        float dt  = dtv[row * DINNER + d];
        float xt  = (float)xs[row * DINNER + d];
        float dtx = dt * xt;
        float y = 0.f;
#pragma unroll
        for (int s = 0; s < DSTATE; ++s) {
            h[s] = __expf(dt * Ad[s]) * h[s] + dtx * sB[s];
            y += h[s] * sC[s];
        }
        y += xt * Dd;
        float z = (float)xz[row * (2 * DINNER) + DINNER + d];
        yg[row * DINNER + d] = (_Float16)(y * silu_f(z));
        __syncthreads();
    }
}

// ---------------- GEMM 4: out = yg @ out_proj^T, stored (B,C,H,W)  (M=8192,N=256,K=512) ----------------

__global__ void gemm_out_kernel(const _Float16* __restrict__ A,
                                const _Float16* __restrict__ W,
                                float* __restrict__ out) {
    const int N = DIM, K = DINNER;
    int wave = threadIdx.x >> 5, lane = threadIdx.x & 31;
    int macro = blockIdx.x * 8 + wave;
    const int mpr = N / 32;                 // 8
    int m0 = (macro / mpr) * 32, n0 = (macro % mpr) * 32;
    const _Float16* A0 = A + (size_t)m0 * K;
    const _Float16* A1 = A + (size_t)(m0 + 16) * K;
    const _Float16* W0 = W + (size_t)n0 * K;
    const _Float16* W1 = W + (size_t)(n0 + 16) * K;
    v8f acc00 = {}, acc01 = {}, acc10 = {}, acc11 = {};
    for (int k0 = 0; k0 < K; k0 += 32) {
        if (k0 + 32 < K) {
            __builtin_prefetch(A0 + (size_t)(lane & 15) * K + k0 + 32, 0, 1);
            __builtin_prefetch(A1 + (size_t)(lane & 15) * K + k0 + 32, 0, 1);
        }
        v16h a0 = load_frag16(A0, K, k0, lane);
        v16h a1 = load_frag16(A1, K, k0, lane);
        v16h b0 = load_frag16(W0, K, k0, lane);
        v16h b1 = load_frag16(W1, K, k0, lane);
        acc00 = wmma32(a0, b0, acc00);
        acc01 = wmma32(a0, b1, acc01);
        acc10 = wmma32(a1, b0, acc10);
        acc11 = wmma32(a1, b1, acc11);
    }
    int g = lane >> 4, n = lane & 15;
#pragma unroll
    for (int r = 0; r < 8; ++r) {
        int mA = m0 + r + 8 * g, mB = m0 + 16 + r + 8 * g;
        int nA = n0 + n, nB = n0 + 16 + n;
        int bA = mA >> 10, lA = mA & 1023;
        int bB = mB >> 10, lB = mB & 1023;
        out[((size_t)(bA * DIM + nA)) * LSEQ + lA] = acc00[r];
        out[((size_t)(bA * DIM + nB)) * LSEQ + lA] = acc01[r];
        out[((size_t)(bB * DIM + nA)) * LSEQ + lB] = acc10[r];
        out[((size_t)(bB * DIM + nB)) * LSEQ + lB] = acc11[r];
    }
}

// ---------------- host launcher ----------------

extern "C" void kernel_launch(void* const* d_in, const int* in_sizes, int n_in,
                              void* d_out, int out_size, void* d_ws, size_t ws_size,
                              hipStream_t stream) {
    const float* x         = (const float*)d_in[0];
    const float* proj_w    = (const float*)d_in[1];
    const float* proj_b    = (const float*)d_in[2];
    const float* in_proj_w = (const float*)d_in[3];
    const float* conv1d_w  = (const float*)d_in[4];
    const float* conv1d_b  = (const float*)d_in[5];
    const float* x_proj_w  = (const float*)d_in[6];
    const float* dt_proj_w = (const float*)d_in[7];
    const float* dt_proj_b = (const float*)d_in[8];
    const float* A_log     = (const float*)d_in[9];
    const float* Dv        = (const float*)d_in[10];
    const float* out_proj_w= (const float*)d_in[11];
    float* out = (float*)d_out;

    char* ws = (char*)d_ws;
    size_t off = 0;
    auto alloc = [&](size_t bytes) { void* p = ws + off; off = (off + bytes + 255) & ~(size_t)255; return p; };
    _Float16* Wf16     = (_Float16*)alloc((size_t)2 * DINNER * DIM * 2);
    float*    bf       = (float*)   alloc((size_t)2 * DINNER * 4);
    _Float16* xseq     = (_Float16*)alloc((size_t)MROWS * DIM * 2);
    _Float16* xz       = (_Float16*)alloc((size_t)MROWS * 2 * DINNER * 2);
    _Float16* xs       = (_Float16*)alloc((size_t)MROWS * DINNER * 2);
    _Float16* dtpad    = (_Float16*)alloc((size_t)MROWS * 32 * 2);
    float*    Bm       = (float*)   alloc((size_t)MROWS * DSTATE * 4);
    float*    Cm       = (float*)   alloc((size_t)MROWS * DSTATE * 4);
    float*    dtv      = (float*)   alloc((size_t)MROWS * DINNER * 4);
    _Float16* yg       = (_Float16*)alloc((size_t)MROWS * DINNER * 2);
    _Float16* xproj16  = (_Float16*)alloc((size_t)48 * DINNER * 2);
    _Float16* dtproj16 = (_Float16*)alloc((size_t)DINNER * 32 * 2);
    _Float16* outproj16= (_Float16*)alloc((size_t)DIM * DINNER * 2);
    float*    Afl      = (float*)   alloc((size_t)DINNER * DSTATE * 4);
    (void)ws_size; (void)in_sizes; (void)n_in; (void)out_size;

    fuse_w_kernel<<<(2 * DINNER * DIM) / 256, 256, 0, stream>>>(
        in_proj_w, proj_w, proj_b, Wf16, bf);
    convert_kernel<<<(CTOT + 255) / 256, 256, 0, stream>>>(
        x_proj_w, dt_proj_w, out_proj_w, A_log, xproj16, dtproj16, outproj16, Afl);
    transpose_x_kernel<<<(BATCH * DIM * LSEQ) / 256, 256, 0, stream>>>(x, xseq);

    // xz = xseq @ Wf^T + bf : 8192x1024, macros = 256*32 = 8192 -> 1024 blocks
    gemm_xz_kernel<<<1024, 256, 0, stream>>>(xseq, Wf16, bf, xz);

    conv_kernel<<<(MROWS * DINNER) / 256, 256, 0, stream>>>(xz, conv1d_w, conv1d_b, xs);

    // x_dbl: 512 m-tiles x 3 n-tiles = 1536 tiles -> 192 blocks
    gemm_xdbl_kernel<<<192, 256, 0, stream>>>(xs, xproj16, dtpad, Bm, Cm);

    // dt: macros = 256*16 = 4096 -> 512 blocks
    gemm_dt_kernel<<<512, 256, 0, stream>>>(dtpad, dtproj16, dt_proj_b, dtv);

    scan_kernel<<<BATCH * 4, 128, 0, stream>>>(dtv, Bm, Cm, xs, xz, Afl, Dv, yg);

    // out: macros = 256*8 = 2048 -> 256 blocks
    gemm_out_kernel<<<256, 256, 0, stream>>>(yg, outproj16, out);
}
